// GCN_75204877353215
// MI455X (gfx1250) — matrix operand order repaired and checked by reading, not compiled, not run on hardware
//
#include <hip/hip_runtime.h>
#include <hip/hip_bf16.h>

typedef __attribute__((ext_vector_type(16))) _Float16 v16h;
typedef __attribute__((ext_vector_type(8)))  float    v8f;

#define TB 256

// ---------------- utility kernels ----------------

__global__ void k_fill(float* __restrict__ p, float v, int n) {
    int i = blockIdx.x * blockDim.x + threadIdx.x;
    if (i < n) p[i] = v;
}

__global__ void k_deg_accum(const long long* __restrict__ dst,
                            float* __restrict__ deg, int e) {
    int i = blockIdx.x * blockDim.x + threadIdx.x;
    if (i < e) atomicAdd(deg + (int)dst[i], 1.0f);
}

__global__ void k_rsqrt_inplace(float* __restrict__ p, int n) {
    int i = blockIdx.x * blockDim.x + threadIdx.x;
    if (i < n) p[i] = rsqrtf(p[i]);   // deg >= 1 always (self loops)
}

// ---------------- WMMA GEMM:  H[nrows x ncols] = X[nrows x 64] @ W[64 x ncols] ----------------
// One wave per 16-row tile; loops over 16-col tiles; K=64 via two 16x16x32 f16 WMMAs.

__global__ void __launch_bounds__(256)
k_gemm_wmma(const float* __restrict__ X, const float* __restrict__ W,
            float* __restrict__ H, int nrows, int ncols) {
    const int lane = threadIdx.x & 31;
    const int wave = threadIdx.x >> 5;
    const int tileRow = blockIdx.x * 8 + wave;
    const int row0 = tileRow << 4;
    if (row0 >= nrows) return;                 // wave-uniform exit: EXEC stays all-ones
    const int m    = lane & 15;
    const int half = lane >> 4;
    int row = row0 + m;
    row = (row < nrows) ? row : (nrows - 1);   // safety clamp (nrows % 16 == 0 in practice)

    // ---- load + convert A (per ISA 16-bit 16x32 A layout: two contiguous 8-runs per lane) ----
    v16h a0, a1;
    {
        const float* xr = X + (size_t)row * 64;
        const float* p = xr + half * 8;        // k-chunk 0: K = half*8 .. +7 and 16+half*8 .. +7
        float4 u = *(const float4*)(p);
        float4 v = *(const float4*)(p + 4);
        float4 s = *(const float4*)(p + 16);
        float4 t = *(const float4*)(p + 20);
        a0[0]=(_Float16)u.x;  a0[1]=(_Float16)u.y;  a0[2]=(_Float16)u.z;  a0[3]=(_Float16)u.w;
        a0[4]=(_Float16)v.x;  a0[5]=(_Float16)v.y;  a0[6]=(_Float16)v.z;  a0[7]=(_Float16)v.w;
        a0[8]=(_Float16)s.x;  a0[9]=(_Float16)s.y;  a0[10]=(_Float16)s.z; a0[11]=(_Float16)s.w;
        a0[12]=(_Float16)t.x; a0[13]=(_Float16)t.y; a0[14]=(_Float16)t.z; a0[15]=(_Float16)t.w;
        const float* q = xr + 32 + half * 8;   // k-chunk 1: K = 32..63
        u = *(const float4*)(q);
        v = *(const float4*)(q + 4);
        s = *(const float4*)(q + 16);
        t = *(const float4*)(q + 20);
        a1[0]=(_Float16)u.x;  a1[1]=(_Float16)u.y;  a1[2]=(_Float16)u.z;  a1[3]=(_Float16)u.w;
        a1[4]=(_Float16)v.x;  a1[5]=(_Float16)v.y;  a1[6]=(_Float16)v.z;  a1[7]=(_Float16)v.w;
        a1[8]=(_Float16)s.x;  a1[9]=(_Float16)s.y;  a1[10]=(_Float16)s.z; a1[11]=(_Float16)s.w;
        a1[12]=(_Float16)t.x; a1[13]=(_Float16)t.y; a1[14]=(_Float16)t.z; a1[15]=(_Float16)t.w;
    }

    const int nct = (ncols + 15) >> 4;
    for (int ct = 0; ct < nct; ++ct) {
        const int col = (ct << 4) + m;
        const bool ok = col < ncols;
        const int colc = ok ? col : (ncols - 1);
        v16h b0, b1;                           // B element e <-> K = half*8 + e (e<8), 16 + half*8 + (e-8) (e>=8)
        #pragma unroll
        for (int e = 0; e < 16; ++e) {
            const int k = (e < 8) ? (half * 8 + e) : (16 + half * 8 + (e - 8));
            float w0 = W[(size_t)k * ncols + colc];
            float w1 = W[(size_t)(k + 32) * ncols + colc];
            b0[e] = (_Float16)(ok ? w0 : 0.0f);
            b1[e] = (_Float16)(ok ? w1 : 0.0f);
        }
        v8f c = {};
        c = __builtin_amdgcn_wmma_f32_16x16x32_f16(false, a0, false, b0, (short)0, c, false, false);
        c = __builtin_amdgcn_wmma_f32_16x16x32_f16(false, a1, false, b1, (short)0, c, false, false);
        if (ok) {
            #pragma unroll
            for (int r = 0; r < 8; ++r)        // D VGPR r <-> row half*8 + r
                H[(size_t)(row0 + half * 8 + r) * ncols + col] = c[r];
        }
    }
}

// ---------------- aggregation ----------------
// agg[i] = bias + dinv[i]^2 * h[i]  (self-loop term, non-atomic), then edge scatter-adds.

template <int CG>   // CG = channels/4 (16 for d=64, 10 for d=40)
__global__ void k_agg_init(const float* __restrict__ h, const float* __restrict__ dinv,
                           const float* __restrict__ bias, float* __restrict__ agg, int n) {
    int i = blockIdx.x * blockDim.x + threadIdx.x;
    if (i >= n * CG) return;
    int row = i / CG, g = i % CG;
    float w = dinv[row]; w *= w;
    float4 hv = *(const float4*)(h + (size_t)row * (CG * 4) + g * 4);
    float4 bv = *(const float4*)(bias + g * 4);
    float4 o;
    o.x = bv.x + w * hv.x;  o.y = bv.y + w * hv.y;
    o.z = bv.z + w * hv.z;  o.w = bv.w + w * hv.w;
    *(float4*)(agg + (size_t)row * (CG * 4) + g * 4) = o;
}

template <int CG>
__global__ void k_agg_edges(const float* __restrict__ h, const float* __restrict__ dinv,
                            const long long* __restrict__ src, const long long* __restrict__ dst,
                            float* __restrict__ agg, int total) {
    int i = blockIdx.x * blockDim.x + threadIdx.x;
    if (i >= total) return;
    int ed = i / CG, g = i % CG;
    int s = (int)src[ed], d = (int)dst[ed];
    float w = dinv[s] * dinv[d];
    float4 hv = *(const float4*)(h + (size_t)s * (CG * 4) + g * 4);
    float* ap = agg + (size_t)d * (CG * 4) + g * 4;
    atomicAdd(ap + 0, w * hv.x);
    atomicAdd(ap + 1, w * hv.y);
    atomicAdd(ap + 2, w * hv.z);
    atomicAdd(ap + 3, w * hv.w);
}

// ---------------- BatchNorm (training stats) over 64 channels ----------------

__global__ void __launch_bounds__(256)
k_bn_partial(const float* __restrict__ x, float* __restrict__ gsum, int n) {
    __shared__ float sh[128];                  // [0..63]=sum, [64..127]=sumsq
    int tid = threadIdx.x;
    if (tid < 128) sh[tid] = 0.0f;
    __syncthreads();
    int g = tid & 15;                          // channel group (4 channels)
    int row = blockIdx.x * (blockDim.x >> 4) + (tid >> 4);
    int stride = gridDim.x * (blockDim.x >> 4);
    float s0=0,s1=0,s2=0,s3=0,q0=0,q1=0,q2=0,q3=0;
    for (; row < n; row += stride) {
        float4 v = *(const float4*)(x + (size_t)row * 64 + g * 4);
        s0 += v.x; s1 += v.y; s2 += v.z; s3 += v.w;
        q0 += v.x*v.x; q1 += v.y*v.y; q2 += v.z*v.z; q3 += v.w*v.w;
    }
    atomicAdd(&sh[g*4+0], s0); atomicAdd(&sh[g*4+1], s1);
    atomicAdd(&sh[g*4+2], s2); atomicAdd(&sh[g*4+3], s3);
    atomicAdd(&sh[64+g*4+0], q0); atomicAdd(&sh[64+g*4+1], q1);
    atomicAdd(&sh[64+g*4+2], q2); atomicAdd(&sh[64+g*4+3], q3);
    __syncthreads();
    if (tid < 128) atomicAdd(&gsum[tid], sh[tid]);
}

__global__ void k_bn_finalize(const float* __restrict__ gsum, const float* __restrict__ gamma,
                              const float* __restrict__ beta, float* __restrict__ scsh, int n) {
    int c = threadIdx.x;
    if (c < 64) {
        float mean = gsum[c] / (float)n;
        float var  = gsum[64 + c] / (float)n - mean * mean;
        float sc   = gamma[c] * rsqrtf(var + 1e-5f);
        scsh[c]      = sc;
        scsh[64 + c] = beta[c] - mean * sc;
    }
}

__global__ void k_bn_apply(float* __restrict__ x, const float* __restrict__ scsh, int n) {
    int i = blockIdx.x * blockDim.x + threadIdx.x;   // over n*16 float4 groups
    if (i >= n * 16) return;
    int row = i >> 4, g = i & 15;
    float4 v  = *(const float4*)(x + (size_t)row * 64 + g * 4);
    float4 sc = *(const float4*)(scsh + g * 4);
    float4 sf = *(const float4*)(scsh + 64 + g * 4);
    v.x = v.x * sc.x + sf.x;  v.y = v.y * sc.y + sf.y;
    v.z = v.z * sc.z + sf.z;  v.w = v.w * sc.w + sf.w;
    *(float4*)(x + (size_t)row * 64 + g * 4) = v;
}

// ---------------- log_softmax: one wave32 per row of C<=64 ----------------

__global__ void __launch_bounds__(256)
k_logsoftmax(const float* __restrict__ emb, float* __restrict__ out, int n, int C) {
    int lane = threadIdx.x & 31;
    int wave = threadIdx.x >> 5;
    int row = blockIdx.x * 8 + wave;
    if (row >= n) return;                      // wave-uniform
    const float* er = emb + (size_t)row * C;
    float e0 = (lane < C)      ? er[lane]      : -3.0e38f;
    float e1 = (lane + 32 < C) ? er[lane + 32] : -3.0e38f;
    float m = fmaxf(e0, e1);
    for (int off = 16; off >= 1; off >>= 1)
        m = fmaxf(m, __shfl_xor(m, off, 32));
    float s = ((lane < C) ? expf(e0 - m) : 0.0f) + ((lane + 32 < C) ? expf(e1 - m) : 0.0f);
    for (int off = 16; off >= 1; off >>= 1)
        s += __shfl_xor(s, off, 32);
    float lse = m + logf(s);
    if (lane < C)      out[(size_t)row * C + lane]      = e0 - lse;
    if (lane + 32 < C) out[(size_t)row * C + lane + 32] = e1 - lse;
}

// ---------------- launcher ----------------

extern "C" void kernel_launch(void* const* d_in, const int* in_sizes, int n_in,
                              void* d_out, int out_size, void* d_ws, size_t ws_size,
                              hipStream_t stream) {
    const float*     x   = (const float*)d_in[0];
    const long long* ei  = (const long long*)d_in[1];
    const float*     W1  = (const float*)d_in[2];
    const float*     b1  = (const float*)d_in[3];
    const float*     W2  = (const float*)d_in[4];
    const float*     b2  = (const float*)d_in[5];
    const float*     W3  = (const float*)d_in[6];
    const float*     b3  = (const float*)d_in[7];
    const float*     g1  = (const float*)d_in[8];
    const float*     be1 = (const float*)d_in[9];
    const float*     g2  = (const float*)d_in[10];
    const float*     be2 = (const float*)d_in[11];

    const int N  = in_sizes[0] / 64;      // 100000
    const int E  = in_sizes[1] / 2;       // 1600000
    const int DO = in_sizes[6] / 64;      // 40

    float* ws   = (float*)d_ws;
    float* dinv = ws;                                 // N
    float* bufH = ws + N;                             // N*64
    float* bufA = bufH + (size_t)N * 64;              // N*64
    float* gsum = bufA + (size_t)N * 64;              // 128
    float* scsh = gsum + 128;                         // 128

    float* outp = (float*)d_out;
    float* emb  = outp + (size_t)N * DO;

    const long long* srcp = ei;
    const long long* dstp = ei + E;

    auto cdiv = [](long long a, long long b) { return (int)((a + b - 1) / b); };
    const int E16 = E * 16;
    const int E10 = E * 10;

    // degree (self-loops folded in as init value 1), then dinv = deg^-1/2
    k_fill<<<cdiv(N, TB), TB, 0, stream>>>(dinv, 1.0f, N);
    k_deg_accum<<<cdiv(E, TB), TB, 0, stream>>>(dstp, dinv, E);
    k_rsqrt_inplace<<<cdiv(N, TB), TB, 0, stream>>>(dinv, N);

    // ---- layer 1 ----
    k_gemm_wmma<<<cdiv(N, 128), TB, 0, stream>>>(x, W1, bufH, N, 64);
    k_agg_init<16><<<cdiv((long long)N * 16, TB), TB, 0, stream>>>(bufH, dinv, b1, bufA, N);
    k_agg_edges<16><<<cdiv(E16, TB), TB, 0, stream>>>(bufH, dinv, srcp, dstp, bufA, E16);
    k_fill<<<1, 128, 0, stream>>>(gsum, 0.0f, 128);
    k_bn_partial<<<512, TB, 0, stream>>>(bufA, gsum, N);
    k_bn_finalize<<<1, 64, 0, stream>>>(gsum, g1, be1, scsh, N);
    k_bn_apply<<<cdiv((long long)N * 16, TB), TB, 0, stream>>>(bufA, scsh, N);

    // ---- layer 2 ----
    k_gemm_wmma<<<cdiv(N, 128), TB, 0, stream>>>(bufA, W2, bufH, N, 64);
    k_agg_init<16><<<cdiv((long long)N * 16, TB), TB, 0, stream>>>(bufH, dinv, b2, bufA, N);
    k_agg_edges<16><<<cdiv(E16, TB), TB, 0, stream>>>(bufH, dinv, srcp, dstp, bufA, E16);
    k_fill<<<1, 128, 0, stream>>>(gsum, 0.0f, 128);
    k_bn_partial<<<512, TB, 0, stream>>>(bufA, gsum, N);
    k_bn_finalize<<<1, 64, 0, stream>>>(gsum, g2, be2, scsh, N);
    k_bn_apply<<<cdiv((long long)N * 16, TB), TB, 0, stream>>>(bufA, scsh, N);

    // ---- layer 3 (emb written directly into second half of d_out) ----
    k_gemm_wmma<<<cdiv(N, 128), TB, 0, stream>>>(bufA, W3, bufH, N, DO);
    k_agg_init<10><<<cdiv((long long)N * 10, TB), TB, 0, stream>>>(bufH, dinv, b3, emb, N);
    k_agg_edges<10><<<cdiv(E10, TB), TB, 0, stream>>>(bufH, dinv, srcp, dstp, emb, E10);

    // log_softmax into first half of d_out
    k_logsoftmax<<<cdiv(N, 8), TB, 0, stream>>>(emb, outp, N, DO);
}
